// PraxisScatter_84439057039459
// MI455X (gfx1250) — matrix operand order repaired and checked by reading, not compiled, not run on hardware
//
#include <hip/hip_runtime.h>
#include <hip/hip_bf16.h>
#include <math.h>

// Problem constants (match reference)
#define B_   8
#define S_   128
#define D_   1024
#define H_   4096
#define KSEL 32768   // min(TOPK*S, S*H)

typedef __bf16 bf16_t;
typedef bf16_t v16bf __attribute__((ext_vector_type(16)));
typedef bf16_t v8bf  __attribute__((ext_vector_type(8)));
typedef float  v8f   __attribute__((ext_vector_type(8)));
typedef unsigned u32x4_t __attribute__((ext_vector_type(4)));
typedef int      i32x8_t __attribute__((ext_vector_type(8)));
typedef int      i32x4_t __attribute__((ext_vector_type(4)));

#ifndef __has_builtin
#define __has_builtin(x) 0
#endif
#if __has_builtin(__builtin_amdgcn_tensor_load_to_lds)
#define HAVE_TDM 1
#else
#define HAVE_TDM 0
#endif

__device__ __forceinline__ bf16_t f2bf(float x) {
  unsigned u = __builtin_bit_cast(unsigned, x);
  u = (u + 0x7FFFu + ((u >> 16) & 1u)) >> 16;   // round-to-nearest-even
  unsigned short s = (unsigned short)u;
  return __builtin_bit_cast(bf16_t, s);
}

// Monotonic float -> uint key (larger key == larger float)
__device__ __forceinline__ unsigned fkey(float x) {
  unsigned u = __builtin_bit_cast(unsigned, x);
  return (u & 0x80000000u) ? ~u : (u | 0x80000000u);
}

// B fragment (K x 16, 16-bit): lane half h holds K = kb+h*16 .. +15, contiguous.
__device__ __forceinline__ v16bf ld_b_frag(const bf16_t* __restrict__ p) {
  const v8bf* q = (const v8bf*)p;
  v8bf a = q[0], b = q[1];
  v16bf f;
#pragma unroll
  for (int i = 0; i < 8; ++i) { f[i] = a[i]; f[i + 8] = b[i]; }
  return f;
}

// A fragment (16 x K, 16-bit): lane half h holds K = kb+h*16 .. +15, contiguous.
__device__ __forceinline__ v16bf ld_a_frag(const bf16_t* __restrict__ row, int kb, int half) {
  const v8bf* q0 = (const v8bf*)(row + kb + half * 16);
  const v8bf* q1 = (const v8bf*)(row + kb + half * 16 + 8);
  v8bf a = *q0, b = *q1;
  v16bf f;
#pragma unroll
  for (int i = 0; i < 8; ++i) { f[i] = a[i]; f[i + 8] = b[i]; }
  return f;
}

__device__ __forceinline__ float apply_act(float v, int act) {
  if (act == 1)      v = fmaxf(v, 0.0f);
  else if (act == 2) v = 0.5f * v * (1.0f + erff(v * 0.70710678118654752f));
  return v;
}

// ---------------------------------------------------------------------------
// TDM-staged bf16 WMMA GEMM: out[M,N] = act(A[M,K] * W[N,K]^T + bias)
// Block tile 128x128 (8 waves), wave tile 64x32 (4x2 wmma accumulators).
// The 128x32 W tile (8KB) is DMA'd into LDS by the Tensor Data Mover
// (double buffered); B fragments are read from LDS, A fragments from global.
// ---------------------------------------------------------------------------
__global__ __launch_bounds__(256) void praxis_gemm_wmma_tdm(
    const bf16_t* __restrict__ A, const bf16_t* __restrict__ W,
    const float* __restrict__ bias,
    float* __restrict__ outF, bf16_t* __restrict__ outB,
    int M, int N, int K, int act)
{
  __shared__ bf16_t ldsW[2][128 * 32];   // [buf][n_local * 32 + k_local]

  const int tid  = threadIdx.x;
  const int lane = tid & 31;
  const int wid  = tid >> 5;
  const int half = lane >> 4;
  const int l16  = lane & 15;

  const int m0    = blockIdx.y * 128;
  const int n0    = blockIdx.x * 128;
  const int mbase = m0 + (wid & 1) * 64;
  const int nbase = n0 + (wid >> 1) * 32;
  const int nloc  = nbase - n0;          // wave's n offset inside the LDS tile

#if HAVE_TDM
  // Issue one TDM descriptor: 2D tile (32 K-elems x 128 N-rows) of the
  // row-major W[N,K] tensor -> packed row-major LDS tile.
  auto tdm_issue = [&](int kb, int buf) {
    unsigned lds = (unsigned)(size_t)(&ldsW[buf][0]);   // low 32 bits = LDS offset
    unsigned long long ga =
        (unsigned long long)(size_t)(W + (size_t)n0 * K + kb);
    u32x4_t g0;
    g0[0] = 1u;                                          // count=1, load, no gather
    g0[1] = lds;                                         // lds_addr
    g0[2] = (unsigned)(ga & 0xFFFFFFFFull);              // global_addr[31:0]
    g0[3] = (unsigned)((ga >> 32) & 0x01FFFFFFull)       // global_addr[56:32]
          | 0x80000000u;                                 // type=2 ("image")
    i32x8_t g1;
    g1[0] = (int)(1u << 16);                             // data_size = 2 bytes
    g1[1] = (int)(((unsigned)K & 0xFFFFu) << 16);        // tensor_dim0[15:0]
    g1[2] = (int)((((unsigned)K >> 16) & 0xFFFFu)        // tensor_dim0[31:16]
          | (((unsigned)N & 0xFFFFu) << 16));            // tensor_dim1[15:0]
    g1[3] = (int)((((unsigned)N >> 16) & 0xFFFFu)        // tensor_dim1[31:16]
          | (32u << 16));                                // tile_dim0 = 32
    g1[4] = (int)128u;                                   // tile_dim1=128, tile_dim2=0
    g1[5] = (int)(unsigned)K;                            // tensor_dim0_stride[31:0]
    g1[6] = 0;                                           // stride hi / dim1_stride
    g1[7] = 0;
    i32x4_t z4 = {0, 0, 0, 0};
#if __clang_major__ >= 23
    i32x8_t z8 = {0, 0, 0, 0, 0, 0, 0, 0};
    __builtin_amdgcn_tensor_load_to_lds(g0, g1, z4, z4, z8, 0);
#else
    __builtin_amdgcn_tensor_load_to_lds(g0, g1, z4, z4, 0);
#endif
  };
#else
  // Fallback staging: cooperative global->LDS copy (8-elem chunks).
  auto stage = [&](int kb, int buf) {
    for (int i = tid; i < (128 * 32) / 8; i += 256) {
      int nl = i >> 2;
      int kc = (i & 3) * 8;
      *(v8bf*)&ldsW[buf][nl * 32 + kc] =
          *(const v8bf*)(W + (size_t)(n0 + nl) * K + kb + kc);
    }
  };
#endif

  const v8f zero = {0.f, 0.f, 0.f, 0.f, 0.f, 0.f, 0.f, 0.f};
  v8f acc[4][2];
#pragma unroll
  for (int mi = 0; mi < 4; ++mi)
#pragma unroll
    for (int ni = 0; ni < 2; ++ni) acc[mi][ni] = zero;

  const bf16_t* arow[4];
#pragma unroll
  for (int mi = 0; mi < 4; ++mi)
    arow[mi] = A + (size_t)(mbase + mi * 16 + l16) * K;

  // preload first tile
#if HAVE_TDM
  if (wid == 0) {
    tdm_issue(0, 0);
    __builtin_amdgcn_s_wait_tensorcnt(0);
  }
  __syncthreads();
#else
  stage(0, 0);
  __syncthreads();
#endif

  int buf = 0;
  for (int kb = 0; kb < K; kb += 32) {
    const bool more = (kb + 32) < K;
#if HAVE_TDM
    if (wid == 0 && more) tdm_issue(kb + 32, buf ^ 1);   // overlap DMA with math
#endif
    v16bf bf0 = ld_b_frag(&ldsW[buf][(nloc +  0 + l16) * 32 + half * 16]);
    v16bf bf1 = ld_b_frag(&ldsW[buf][(nloc + 16 + l16) * 32 + half * 16]);
    __builtin_prefetch(arow[0] + kb + 64, 0, 1);
#pragma unroll
    for (int mi = 0; mi < 4; ++mi) {
      v16bf af = ld_a_frag(arow[mi], kb, half);
      acc[mi][0] = __builtin_amdgcn_wmma_f32_16x16x32_bf16(
          false, af, false, bf0, (short)0, acc[mi][0], false, false);
      acc[mi][1] = __builtin_amdgcn_wmma_f32_16x16x32_bf16(
          false, af, false, bf1, (short)0, acc[mi][1], false, false);
    }
#if HAVE_TDM
    if (wid == 0 && more) __builtin_amdgcn_s_wait_tensorcnt(0);
    __syncthreads();
#else
    __syncthreads();
    if (more) stage(kb + 32, buf ^ 1);
    __syncthreads();
#endif
    buf ^= 1;
  }

#pragma unroll
  for (int ni = 0; ni < 2; ++ni) {
    int col = nbase + ni * 16 + l16;
    float bv = bias[col];
#pragma unroll
    for (int mi = 0; mi < 4; ++mi) {
      v8f c = acc[mi][ni];
#pragma unroll
      for (int r = 0; r < 8; ++r) {
        float v = apply_act(c[r] + bv, act);
        int row = mbase + mi * 16 + half * 8 + r;
        size_t o = (size_t)row * N + col;
        if (outF) outF[o] = v;
        else      outB[o] = f2bf(v);
      }
    }
  }
}

// ---------------------------------------------------------------------------
// Direct-load WMMA GEMM with per-(batch,row) weight/bias selection
// (sel!=0 -> Walt/biasAlt). Used for the row-scatter GEMM where each B-lane
// needs its own weight-matrix base pointer (not expressible as one TDM tile).
// ---------------------------------------------------------------------------
__global__ __launch_bounds__(256) void praxis_gemm_wmma_sel(
    const bf16_t* __restrict__ A, const bf16_t* __restrict__ W,
    const bf16_t* __restrict__ Walt, const int* __restrict__ selMask,
    const float* __restrict__ bias, const float* __restrict__ biasAlt,
    bf16_t* __restrict__ outB,
    int M, int N, int K, int act)
{
  const int tid  = threadIdx.x;
  const int lane = tid & 31;
  const int wid  = tid >> 5;
  const int half = lane >> 4;
  const int l16  = lane & 15;

  const int m0    = blockIdx.y * 128;
  const int n0    = blockIdx.x * 128;
  const int mbase = m0 + (wid & 1) * 64;
  const int nbase = n0 + (wid >> 1) * 32;
  const int batch = m0 >> 7;           // 128 rows per batch element (S_ == 128)

  const v8f zero = {0.f, 0.f, 0.f, 0.f, 0.f, 0.f, 0.f, 0.f};
  v8f acc[4][2];
#pragma unroll
  for (int mi = 0; mi < 4; ++mi)
#pragma unroll
    for (int ni = 0; ni < 2; ++ni) acc[mi][ni] = zero;

  const bf16_t* wrow[2];
#pragma unroll
  for (int ni = 0; ni < 2; ++ni) {
    int nr = nbase + ni * 16 + l16;
    wrow[ni] = (selMask[batch * N + nr] ? Walt : W) + (size_t)nr * K;
  }

  const bf16_t* arow[4];
#pragma unroll
  for (int mi = 0; mi < 4; ++mi)
    arow[mi] = A + (size_t)(mbase + mi * 16 + l16) * K;

  for (int kb = 0; kb < K; kb += 32) {
    v16bf bf0 = ld_b_frag(wrow[0] + kb + half * 16);
    v16bf bf1 = ld_b_frag(wrow[1] + kb + half * 16);
    __builtin_prefetch(wrow[0] + kb + 64, 0, 1);
    __builtin_prefetch(arow[0] + kb + 64, 0, 1);
#pragma unroll
    for (int mi = 0; mi < 4; ++mi) {
      v16bf af = ld_a_frag(arow[mi], kb, half);
      acc[mi][0] = __builtin_amdgcn_wmma_f32_16x16x32_bf16(
          false, af, false, bf0, (short)0, acc[mi][0], false, false);
      acc[mi][1] = __builtin_amdgcn_wmma_f32_16x16x32_bf16(
          false, af, false, bf1, (short)0, acc[mi][1], false, false);
    }
  }

#pragma unroll
  for (int ni = 0; ni < 2; ++ni) {
    int col = nbase + ni * 16 + l16;
    float bv = selMask[batch * N + col] ? biasAlt[col] : bias[col];
#pragma unroll
    for (int mi = 0; mi < 4; ++mi) {
      v8f c = acc[mi][ni];
#pragma unroll
      for (int r = 0; r < 8; ++r) {
        float v = apply_act(c[r] + bv, act);
        int row = mbase + mi * 16 + half * 8 + r;
        outB[(size_t)row * N + col] = f2bf(v);
      }
    }
  }
}

// ---------------------------------------------------------------------------
// fp32 -> bf16 conversion
// ---------------------------------------------------------------------------
__global__ void praxis_cvt_bf16(const float* __restrict__ in, bf16_t* __restrict__ out, int n) {
  int i = blockIdx.x * blockDim.x + threadIdx.x;
  int stride = gridDim.x * blockDim.x;
  for (; i < n; i += stride) out[i] = f2bf(in[i]);
}

// ---------------------------------------------------------------------------
// Two-level radix-histogram top-k threshold (per batch, 2048 bins per level)
// ---------------------------------------------------------------------------
__global__ void praxis_hist_hi(const float* __restrict__ scores, unsigned* __restrict__ hist) {
  const int b = blockIdx.y;
  const float* s = scores + (size_t)b * (S_ * H_);
  int i = blockIdx.x * blockDim.x + threadIdx.x;
  int stride = gridDim.x * blockDim.x;
  for (; i < S_ * H_; i += stride)
    atomicAdd(&hist[b * 2048 + (fkey(s[i]) >> 21)], 1u);
}

__global__ void praxis_scan_hi(const unsigned* __restrict__ hist,
                               unsigned* __restrict__ bin1, unsigned* __restrict__ rem) {
  int b = threadIdx.x;
  if (b >= B_) return;
  unsigned acc = 0;
  for (int bin = 2047; bin >= 0; --bin) {
    unsigned c = hist[b * 2048 + bin];
    if (acc + c >= (unsigned)KSEL) { bin1[b] = (unsigned)bin; rem[b] = (unsigned)KSEL - acc; return; }
    acc += c;
  }
  bin1[b] = 0; rem[b] = 1;
}

__global__ void praxis_hist_lo(const float* __restrict__ scores,
                               const unsigned* __restrict__ bin1, unsigned* __restrict__ hist) {
  const int b = blockIdx.y;
  const unsigned target = bin1[b];
  const float* s = scores + (size_t)b * (S_ * H_);
  int i = blockIdx.x * blockDim.x + threadIdx.x;
  int stride = gridDim.x * blockDim.x;
  for (; i < S_ * H_; i += stride) {
    unsigned k = fkey(s[i]);
    if ((k >> 21) == target)
      atomicAdd(&hist[b * 2048 + ((k >> 10) & 2047u)], 1u);
  }
}

__global__ void praxis_scan_lo(const unsigned* __restrict__ hist, const unsigned* __restrict__ bin1,
                               const unsigned* __restrict__ rem, unsigned* __restrict__ thr) {
  int b = threadIdx.x;
  if (b >= B_) return;
  unsigned need = rem[b];
  unsigned acc = 0;
  unsigned bin2 = 0;
  for (int bin = 2047; bin >= 0; --bin) {
    unsigned c = hist[b * 2048 + bin];
    if (acc + c >= need) { bin2 = (unsigned)bin; break; }
    acc += c;
  }
  thr[b] = (bin1[b] << 21) | (bin2 << 10);
}

// sel[b][h] = any_s ( scores[b,s,h] >= threshold[b] )
__global__ void praxis_mark_sel(const float* __restrict__ scores,
                                const unsigned* __restrict__ thr, int* __restrict__ sel) {
  int t = blockIdx.x * blockDim.x + threadIdx.x;
  if (t >= B_ * H_) return;
  int b = t / H_, h = t - b * H_;
  unsigned th = thr[b];
  int f = 0;
  for (int s = 0; s < S_; ++s)
    f |= (fkey(scores[((size_t)(b * S_ + s)) * H_ + h]) >= th) ? 1 : 0;
  sel[t] = f;
}

// ---------------------------------------------------------------------------
extern "C" void kernel_launch(void* const* d_in, const int* in_sizes, int n_in,
                              void* d_out, int out_size, void* d_ws, size_t ws_size,
                              hipStream_t stream)
{
  const float* x         = (const float*)d_in[0];
  const float* up_prev_w = (const float*)d_in[1];
  const float* up_prev_b = (const float*)d_in[2];
  const float* up_curr_w = (const float*)d_in[3];
  const float* up_curr_b = (const float*)d_in[4];
  const float* gate_w1   = (const float*)d_in[5];
  const float* gate_b1   = (const float*)d_in[6];
  const float* gate_w2   = (const float*)d_in[7];
  const float* gate_b2   = (const float*)d_in[8];
  const float* down_w    = (const float*)d_in[9];
  const float* down_b    = (const float*)d_in[10];
  (void)in_sizes; (void)n_in; (void)out_size; (void)ws_size;

  char* ws = (char*)d_ws;
  size_t off = 0;
  auto take = [&](size_t bytes) -> char* {
    char* p = ws + off;
    off += (bytes + 255) & ~(size_t)255;
    return p;
  };

  bf16_t*   xbf    = (bf16_t*)  take((size_t)B_ * S_ * D_ * 2);
  bf16_t*   w1bf   = (bf16_t*)  take((size_t)H_ * D_ * 2);
  bf16_t*   w2bf   = (bf16_t*)  take((size_t)H_ * H_ * 2);
  bf16_t*   wpbf   = (bf16_t*)  take((size_t)H_ * D_ * 2);
  bf16_t*   wcbf   = (bf16_t*)  take((size_t)H_ * D_ * 2);
  bf16_t*   wdbf   = (bf16_t*)  take((size_t)D_ * H_ * 2);
  bf16_t*   gbuf   = (bf16_t*)  take((size_t)B_ * S_ * H_ * 2);
  float*    scores = (float*)   take((size_t)B_ * S_ * H_ * 4);
  bf16_t*   hbuf   = (bf16_t*)  take((size_t)B_ * S_ * H_ * 2);
  unsigned* hist1  = (unsigned*)take((size_t)B_ * 2048 * 4);
  unsigned* hist2  = (unsigned*)take((size_t)B_ * 2048 * 4);
  unsigned* bin1   = (unsigned*)take((size_t)B_ * 4);
  unsigned* rem    = (unsigned*)take((size_t)B_ * 4);
  unsigned* thr    = (unsigned*)take((size_t)B_ * 4);
  int*      sel    = (int*)     take((size_t)B_ * H_ * 4);

  const dim3 cb(256);
  // fp32 -> bf16 for activations and all weight matrices
  praxis_cvt_bf16<<<1024, cb, 0, stream>>>(x,         xbf,  B_ * S_ * D_);
  praxis_cvt_bf16<<<2048, cb, 0, stream>>>(gate_w1,   w1bf, H_ * D_);
  praxis_cvt_bf16<<<4096, cb, 0, stream>>>(gate_w2,   w2bf, H_ * H_);
  praxis_cvt_bf16<<<2048, cb, 0, stream>>>(up_prev_w, wpbf, H_ * D_);
  praxis_cvt_bf16<<<2048, cb, 0, stream>>>(up_curr_w, wcbf, H_ * D_);
  praxis_cvt_bf16<<<2048, cb, 0, stream>>>(down_w,    wdbf, D_ * H_);

  const dim3 blk(256);
  // g = relu(x @ W1^T + b1)          [1024 x 4096, K=1024]
  praxis_gemm_wmma_tdm<<<dim3(H_ / 128, (B_ * S_) / 128), blk, 0, stream>>>(
      xbf, w1bf, gate_b1, nullptr, gbuf, B_ * S_, H_, D_, 1);
  // scores = g @ W2^T + b2           [1024 x 4096, K=4096] -> f32
  praxis_gemm_wmma_tdm<<<dim3(H_ / 128, (B_ * S_) / 128), blk, 0, stream>>>(
      gbuf, w2bf, gate_b2, scores, nullptr, B_ * S_, H_, H_, 0);

  // per-batch top-k threshold + hidden-row selection mask
  hipMemsetAsync(hist1, 0, (size_t)B_ * 2048 * 4, stream);
  hipMemsetAsync(hist2, 0, (size_t)B_ * 2048 * 4, stream);
  praxis_hist_hi <<<dim3(256, B_), cb, 0, stream>>>(scores, hist1);
  praxis_scan_hi <<<1, 32, 0, stream>>>(hist1, bin1, rem);
  praxis_hist_lo <<<dim3(256, B_), cb, 0, stream>>>(scores, bin1, hist2);
  praxis_scan_lo <<<1, 32, 0, stream>>>(hist2, bin1, rem, thr);
  praxis_mark_sel<<<(B_ * H_) / 256, cb, 0, stream>>>(scores, thr, sel);

  // h = gelu(x @ mod_W^T + mod_b)    row-scatter between up_curr / up_prev
  praxis_gemm_wmma_sel<<<dim3(H_ / 128, (B_ * S_) / 128), blk, 0, stream>>>(
      xbf, wcbf, wpbf, sel, up_curr_b, up_prev_b, hbuf, B_ * S_, H_, D_, 2);
  // out = h @ down_w^T + down_b      [1024 x 1024, K=4096] -> f32 d_out
  praxis_gemm_wmma_tdm<<<dim3(D_ / 128, (B_ * S_) / 128), blk, 0, stream>>>(
      hbuf, wdbf, down_b, (float*)d_out, nullptr, B_ * S_, D_, H_, 0);
}
